// ImplicitNeuralFunction_197568496093
// MI455X (gfx1250) — matrix operand order repaired and checked by reading, not compiled, not run on hardware
//
#include <hip/hip_runtime.h>
#include <hip/hip_bf16.h>

typedef __attribute__((ext_vector_type(16))) _Float16 v16h;
typedef __attribute__((ext_vector_type(8)))  float    v8f;

union AF {
    v16h  v;
    uint4 q[2];
    _Float16 h[16];
};

#define ASTR 264   // LDS activation row stride in halves (264*2B = 528B = 33*16B, bank-skewed)

// ---------------------------------------------------------------------------
// WMMA helpers (CDNA5 16x16x32 f16 -> f32)
// ---------------------------------------------------------------------------
__device__ __forceinline__ v8f wmma_f16(v16h a, v16h b, v8f c) {
    return __builtin_amdgcn_wmma_f32_16x16x32_f16(
        /*neg_a=*/false, a, /*neg_b=*/false, b,
        /*c_mod=*/(short)0, c, /*reuse_a=*/false, /*reuse_b=*/false);
}

__device__ __forceinline__ v8f splat8(float v) {
    return (v8f){v, v, v, v, v, v, v, v};
}

// Fast silu: x * rcp(1+exp(-x)) -> v_exp_f32 + v_rcp_f32, no div_scale chain
__device__ __forceinline__ float silu(float v) {
    return v * __builtin_amdgcn_rcpf(1.0f + __expf(-v));
}

// A-matrix fragment (16x32 f16, M=points): row-major [row][stride] source.
// lanes 0-15: M=lane, K = k0+0..15
// lanes 16-31: M=lane-16, K = k0+16..31
__device__ __forceinline__ v16h load_a(const _Float16* base, int stride, int lane, int k0) {
    int r   = lane & 15;
    int off = (lane >> 4) * 16;
    const _Float16* p = base + r * stride + k0 + off;
    AF f;
    f.q[0] = *(const uint4*)(p);
    f.q[1] = *(const uint4*)(p + 8);
    return f.v;
}

// B-matrix fragment (32x16 f16): B[k][n] = W[n][k], W row-major [out][Kpad].
// lane holds column n = nt*16 + (lane&15); K range 0-15 (lanes 0-15) / 16-31 (lanes 16-31).
__device__ __forceinline__ v16h load_bf(const _Float16* W, int stride, int lane, int nt, int k0) {
    int n  = nt * 16 + (lane & 15);
    int ks = k0 + (lane >> 4) * 16;
    const _Float16* p = W + (size_t)n * stride + ks;
    AF f;
    f.q[0] = *(const uint4*)(p);
    f.q[1] = *(const uint4*)(p + 8);
    return f.v;
}

// C/D layout: VGPR j -> row ((lane>=16)?8:0)+j, col = nt*16 + (lane&15).
// Apply silu and store back to LDS activation tile as f16.
__device__ __forceinline__ void store_silu(_Float16 (*act)[ASTR], int lane, int nt, v8f c) {
    int col = nt * 16 + (lane & 15);
    int rb  = (lane >> 4) * 8;
#pragma unroll
    for (int j = 0; j < 8; ++j) {
        act[rb + j][col] = (_Float16)silu(c[j]);
    }
}

// ---------------------------------------------------------------------------
// Kernel 1: NeRF positional encoding -> f16 [N][64], cols 62,63 zero.
// h[0:2]=x, h[2+2s+d]=sin(x_d*2^s), h[32+2s+d]=cos(x_d*2^s), s=0..14
// ---------------------------------------------------------------------------
__global__ __launch_bounds__(128) void encode_kernel(const float* __restrict__ x,
                                                     _Float16* __restrict__ h0, int N) {
    int p = blockIdx.x * blockDim.x + threadIdx.x;
    if (p >= N) return;
    float x0 = x[2 * p + 0];
    float x1 = x[2 * p + 1];
    _Float16 row[64];
    row[0] = (_Float16)x0;
    row[1] = (_Float16)x1;
    float s = 1.0f;
#pragma unroll
    for (int j = 0; j < 15; ++j) {
        float t0 = x0 * s, t1 = x1 * s;
        row[2  + 2 * j + 0] = (_Float16)sinf(t0);
        row[2  + 2 * j + 1] = (_Float16)sinf(t1);
        row[32 + 2 * j + 0] = (_Float16)cosf(t0);
        row[32 + 2 * j + 1] = (_Float16)cosf(t1);
        s *= 2.0f;
    }
    row[62] = (_Float16)0.0f;
    row[63] = (_Float16)0.0f;
    uint4* dst = (uint4*)(h0 + (size_t)p * 64);
    const uint4* src = (const uint4*)row;
#pragma unroll
    for (int i = 0; i < 8; ++i) dst[i] = src[i];
}

// ---------------------------------------------------------------------------
// Kernel 2: hyper weights: w[b,v,c] = dot(latent[b, layer*30+v, :], pw[v,c,:]) + pb
// then layernorm over c, store f16 padded [32][KP] per (b,layer); pad = 0.
// ---------------------------------------------------------------------------
__global__ __launch_bounds__(128) void hyper_kernel(const float* __restrict__ latent,
                                                    const float* __restrict__ pw,
                                                    const float* __restrict__ pb,
                                                    const float* __restrict__ lw,
                                                    const float* __restrict__ lb,
                                                    _Float16* __restrict__ hw,
                                                    int layer, int C, int KP) {
    int b = blockIdx.x;
    __shared__ float w[30][64];
    const float* L = latent + ((size_t)b * 120 + layer * 30) * 120;
    for (int idx = threadIdx.x; idx < 30 * C; idx += blockDim.x) {
        int v = idx / C, c = idx % C;
        const float* sl  = L + (size_t)v * 120;
        const float* pwv = pw + ((size_t)v * C + c) * 120;
        float acc = 0.0f;
        for (int d = 0; d < 120; ++d) acc += sl[d] * pwv[d];
        w[v][c] = acc + pb[v * C + c];
    }
    __syncthreads();
    _Float16* out = hw + (size_t)b * 5120 + ((layer == 0) ? 0 : (2048 + (layer - 1) * 1024));
    for (int v = threadIdx.x; v < 32; v += blockDim.x) {
        if (v < 30) {
            float mu = 0.0f;
            for (int c = 0; c < C; ++c) mu += w[v][c];
            mu /= (float)C;
            float var = 0.0f;
            for (int c = 0; c < C; ++c) { float dv = w[v][c] - mu; var += dv * dv; }
            var /= (float)C;
            float rs = rsqrtf(var + 1e-5f);
            for (int c = 0; c < KP; ++c) {
                float val = (c < C) ? ((w[v][c] - mu) * rs * lw[c] + lb[c]) : 0.0f;
                out[(size_t)v * KP + c] = (_Float16)val;
            }
        } else {
            for (int c = 0; c < KP; ++c) out[(size_t)v * KP + c] = (_Float16)0.0f;
        }
    }
}

// ---------------------------------------------------------------------------
// Kernel 3: convert shared MLP weights to f16 (padded where needed)
// m0: [256][32] from mlp_w0 (256,30); m1: [128][256]; m2: [128] from (1,128)
// ---------------------------------------------------------------------------
__global__ __launch_bounds__(256) void cvt_kernel(const float* __restrict__ w0,
                                                  const float* __restrict__ w1,
                                                  const float* __restrict__ w2,
                                                  _Float16* __restrict__ m0,
                                                  _Float16* __restrict__ m1,
                                                  _Float16* __restrict__ m2) {
    int t = blockIdx.x * blockDim.x + threadIdx.x;
    int stride = gridDim.x * blockDim.x;
    for (int i = t; i < 256 * 32; i += stride) {
        int o = i >> 5, k = i & 31;
        m0[i] = (_Float16)((k < 30) ? w0[o * 30 + k] : 0.0f);
    }
    for (int i = t; i < 128 * 256; i += stride) m1[i] = (_Float16)w1[i];
    for (int i = t; i < 128; i += stride)       m2[i] = (_Float16)w2[i];
}

// ---------------------------------------------------------------------------
// Kernel 4: fused per-point pipeline. One wave = 16 points of one batch.
// 62->30 -> 30->30 x3 (per-batch weights) -> 30->256 -> 256->128 -> 128->1
// 94 v_wmma_f32_16x16x32_f16 per wave.
// ---------------------------------------------------------------------------
__global__ __launch_bounds__(128) void fused_mlp_kernel(
    const _Float16* __restrict__ h0, const _Float16* __restrict__ hw,
    const _Float16* __restrict__ m0, const _Float16* __restrict__ m1,
    const _Float16* __restrict__ m2,
    const float* __restrict__ b0, const float* __restrict__ b1,
    const float* __restrict__ b2, const float* __restrict__ b3,
    const float* __restrict__ mb0, const float* __restrict__ mb1,
    const float* __restrict__ mb2,
    float* __restrict__ out, int N) {
    __shared__ __align__(16) _Float16 s_act[4][16][ASTR];

    const int lane  = threadIdx.x & 31;
    const int wv    = threadIdx.x >> 5;
    const int cn    = lane & 15;
    const int hi    = lane >> 4;
    const int batch = blockIdx.y;
    const int pbase = (blockIdx.x * 4 + wv) * 16;

    _Float16 (*act)[ASTR] = s_act[wv];
    const _Float16* hwb = hw + (size_t)batch * 5120;

    // Warm the hot 64KB w1 matrix into cache (global_prefetch_b8)
    __builtin_prefetch(m1 + (size_t)threadIdx.x * 256, 0, 3);

    // ---- layer 0: K=64 (A from global encoded features), out 30 (pad 32) ----
    v8f c0, c1;
    {
        float bva = b0[cn];
        float bvb = (cn + 16 < 30) ? b0[cn + 16] : 0.0f;
        c0 = splat8(bva);
        c1 = splat8(bvb);
    }
    const _Float16* arow = h0 + (size_t)pbase * 64;
#pragma unroll
    for (int k0 = 0; k0 < 64; k0 += 32) {
        v16h a = load_a(arow, 64, lane, k0);
        c0 = wmma_f16(a, load_bf(hwb, 64, lane, 0, k0), c0);
        c1 = wmma_f16(a, load_bf(hwb, 64, lane, 1, k0), c1);
    }
    store_silu(act, lane, 0, c0);
    store_silu(act, lane, 1, c1);

    // ---- layers 1..3: K=32, out 30 (pad 32), per-batch weights ----
    // A fragment held in registers, so in-place LDS overwrite is safe.
    const _Float16* hwl[3] = { hwb + 2048, hwb + 3072, hwb + 4096 };
    const float*    bil[3] = { b1, b2, b3 };
#pragma unroll
    for (int L = 0; L < 3; ++L) {
        float bva = bil[L][cn];
        float bvb = (cn + 16 < 30) ? bil[L][cn + 16] : 0.0f;
        c0 = splat8(bva);
        c1 = splat8(bvb);
        v16h a = load_a(&act[0][0], ASTR, lane, 0);
        c0 = wmma_f16(a, load_bf(hwl[L], 32, lane, 0, 0), c0);
        c1 = wmma_f16(a, load_bf(hwl[L], 32, lane, 1, 0), c1);
        store_silu(act, lane, 0, c0);
        store_silu(act, lane, 1, c1);
    }

    // ---- MLP0: K=32, out 256 (16 n-tiles), streamed: 1 live accumulator ----
    {
        v16h a = load_a(&act[0][0], ASTR, lane, 0);
#pragma unroll
        for (int nt = 0; nt < 16; ++nt) {
            v8f c = splat8(mb0[nt * 16 + cn]);
            c = wmma_f16(a, load_bf(m0, 32, lane, nt, 0), c);
            store_silu(act, lane, nt, c);
        }
    }

    // ---- MLP1: K=256 (8 k-steps), out 128 (8 n-tiles) -> 64 WMMAs ----
    // Output cols 0..127 overlap input cols 0..255: hold 8 accumulators.
    {
        v8f dd[8];
#pragma unroll
        for (int nt = 0; nt < 8; ++nt) dd[nt] = splat8(mb1[nt * 16 + cn]);
#pragma unroll
        for (int k0 = 0; k0 < 256; k0 += 32) {
            v16h a = load_a(&act[0][0], ASTR, lane, k0);
#pragma unroll
            for (int nt = 0; nt < 8; ++nt)
                dd[nt] = wmma_f16(a, load_bf(m1, 256, lane, nt, k0), dd[nt]);
        }
#pragma unroll
        for (int nt = 0; nt < 8; ++nt) store_silu(act, lane, nt, dd[nt]);
    }

    // ---- MLP2: K=128, out 1 (only column 0 of B non-zero) ----
    {
        v8f e = splat8((cn == 0) ? mb2[0] : 0.0f);
#pragma unroll
        for (int k0 = 0; k0 < 128; k0 += 32) {
            v16h a = load_a(&act[0][0], ASTR, lane, k0);
            AF bz;
            bz.q[0] = make_uint4(0, 0, 0, 0);
            bz.q[1] = make_uint4(0, 0, 0, 0);
            if (cn == 0) {
                const _Float16* p = m2 + k0 + hi * 16;
                bz.q[0] = *(const uint4*)(p);
                bz.q[1] = *(const uint4*)(p + 8);
            }
            e = wmma_f16(a, bz.v, e);
        }
        if (cn == 0) {
            float* o = out + (size_t)batch * N + pbase + hi * 8;
#pragma unroll
            for (int j = 0; j < 8; ++j) o[j] = e[j];
        }
    }
}

// ---------------------------------------------------------------------------
// Host launcher
// ---------------------------------------------------------------------------
extern "C" void kernel_launch(void* const* d_in, const int* in_sizes, int n_in,
                              void* d_out, int out_size, void* d_ws, size_t ws_size,
                              hipStream_t stream) {
    const float* x   = (const float*)d_in[0];
    const float* sdf = (const float*)d_in[1];
    const float* pw[4];  const float* pb[4];
    const float* lw[4];  const float* lb[4];  const float* bi[4];
    for (int i = 0; i < 4; ++i) {
        pw[i] = (const float*)d_in[2 + 5 * i];
        pb[i] = (const float*)d_in[3 + 5 * i];
        lw[i] = (const float*)d_in[4 + 5 * i];
        lb[i] = (const float*)d_in[5 + 5 * i];
        bi[i] = (const float*)d_in[6 + 5 * i];
    }
    const float* mw0f = (const float*)d_in[22];
    const float* mb0  = (const float*)d_in[23];
    const float* mw1f = (const float*)d_in[24];
    const float* mb1  = (const float*)d_in[25];
    const float* mw2f = (const float*)d_in[26];
    const float* mb2  = (const float*)d_in[27];

    const int N = in_sizes[0] / 2;                 // 32768
    const int B = in_sizes[1] / (120 * 120);       // 16

    // Workspace layout (f16 halves)
    _Float16* ws  = (_Float16*)d_ws;
    _Float16* h0  = ws;                            // N*64
    _Float16* hwp = h0 + (size_t)N * 64;           // B*5120
    _Float16* m0  = hwp + (size_t)B * 5120;        // 256*32
    _Float16* m1  = m0 + 256 * 32;                 // 128*256
    _Float16* m2  = m1 + 128 * 256;                // 128

    encode_kernel<<<(N + 127) / 128, 128, 0, stream>>>(x, h0, N);

    const int CH[4] = {62, 30, 30, 30};
    const int KP[4] = {64, 32, 32, 32};
    for (int i = 0; i < 4; ++i)
        hyper_kernel<<<B, 128, 0, stream>>>(sdf, pw[i], pb[i], lw[i], lb[i], hwp, i, CH[i], KP[i]);

    cvt_kernel<<<32, 256, 0, stream>>>(mw0f, mw1f, mw2f, m0, m1, m2);

    dim3 grid(N / 64, B);
    fused_mlp_kernel<<<grid, 128, 0, stream>>>(h0, hwp, m0, m1, m2,
                                               bi[0], bi[1], bi[2], bi[3],
                                               mb0, mb1, mb2,
                                               (float*)d_out, N);
}
